// LUTFR_83614423318794
// MI455X (gfx1250) — compile-verified, matrix operand order checked by reading, not run on hardware
//
#include <hip/hip_runtime.h>
#include <hip/hip_fp16.h>

typedef __attribute__((ext_vector_type(2)))  float    v2f;
typedef __attribute__((ext_vector_type(8)))  float    v8f;
typedef __attribute__((ext_vector_type(16))) _Float16 v16h;

#define LUT_D     33
#define LUT_S     (LUT_D * LUT_D * LUT_D)   // 35937 spatial entries
#define NLUT      350
#define NLUT_K4   (NLUT & ~3)               // 348: K handled by guard-free WMMA loop
#define NCOLS     (3 * LUT_S)               // 107811 columns per lut bank
#define TOTCOLS   (2 * NCOLS)               // 215622 columns across both banks
#define CL_STRIDE (LUT_S * 4)               // floats per collapsed interleaved LUT

// ---------------------------------------------------------------------------
// Kernel 1: CL[l] = sum_n lut[l][n]  via WMMA (ones-matrix x lut-tile).
// D = A(16x4, all ones) x B(4x16, lut tile) + C  =>  every row of D holds the
// column sums of B; accumulate C over K. 87 guard-free iterations cover
// K=0..347; the K=348,349 tail is folded in as a scalar add at store time
// (we only consume D row 0, and D[0][n] = sum_k B[k][n]).
// Output re-laid out channel-interleaved [l][s][4] for B128 corner gathers.
// ---------------------------------------------------------------------------
__global__ __launch_bounds__(256) void lut_reduce_wmma(
    const float* __restrict__ lut, float* __restrict__ cl)
{
    const int lane = threadIdx.x & 31;
    const int wave = blockIdx.x * (blockDim.x >> 5) + (threadIdx.x >> 5);
    const long long e0 = (long long)wave * 16;
    if (e0 >= TOTCOLS) return;                 // wave-uniform exit

    const int  col   = (int)e0 + (lane & 15);
    const bool colOK = col < TOTCOLS;
    const int  colC  = colOK ? col : (TOTCOLS - 1);  // clamp for safe addressing
    const int  l     = colC / NCOLS;
    const int  e     = colC - l * NCOLS;
    const long long srcBase = (long long)l * NLUT * NCOLS + e;

    v8f c = {};

#if __has_builtin(__builtin_amdgcn_wmma_f32_16x16x4_f32)
    // B 4x16 f32 layout -> lanes 0-15: K=kk,kk+1 ; lanes 16-31: K=kk+2,kk+3
    const int krow = (lane >> 4) * 2;
    const v2f a = {1.0f, 1.0f};
    const float* p = lut + srcBase + (long long)krow * NCOLS;
    for (int kk = 0; kk < NLUT_K4; kk += 4) {
        v2f b;
        b.x = p[0];
        b.y = p[NCOLS];
        p += 4LL * NCOLS;
        c = __builtin_amdgcn_wmma_f32_16x16x4_f32(
            /*neg_a=*/false, a, /*neg_b=*/false, b,
            /*c_mod=*/(short)0, c, /*reuse_a=*/false, /*reuse_b=*/false);
    }
#else
    // Fallback: confirmed f16 WMMA, f32 accumulate. B 32x16 16-bit layout:
    // half i of lane -> K = kk + (i<8 ? i : i+8) + (lane>=16 ? 8 : 0)
    const int khalf = (lane >> 4) * 8;
    v16h a;
#pragma unroll
    for (int i = 0; i < 16; ++i) a[i] = (_Float16)1.0f;
    for (int kk = 0; kk + 32 <= NLUT_K4; kk += 32) {
        v16h b;
#pragma unroll
        for (int i = 0; i < 16; ++i) {
            const int K = kk + ((i < 8) ? i : i + 8) + khalf;
            b[i] = (_Float16)lut[srcBase + (long long)K * NCOLS];
        }
        c = __builtin_amdgcn_wmma_f32_16x16x32_f16(
            false, a, false, b, (short)0, c, false, false);
    }
    // f16 path handles K = 320..347 in the scalar tail below
    #undef  NLUT_K4_EFF
    #define NLUT_K4_EFF 320
#endif

#ifndef NLUT_K4_EFF
#define NLUT_K4_EFF NLUT_K4
#endif
    // Scalar tail: remaining K rows added directly (row-0 equivalence).
    float tail = 0.0f;
    for (int K = NLUT_K4_EFF; K < NLUT; ++K)
        tail += lut[srcBase + (long long)K * NCOLS];

    // D row 0 lives in VGPR0 (c[0]) of lanes 0..15.
    if ((lane < 16) && colOK) {
        const int ch = e / LUT_S;
        const int s  = e - ch * LUT_S;
        cl[(long long)l * CL_STRIDE + (long long)s * 4 + ch] = c[0] + tail;
    }
}

// ---------------------------------------------------------------------------
// Kernel 2: fused double trilinear apply. One thread = one pixel.
// Collapsed LUTs are [33][33][33][4] f32 -> each corner is one B128 load
// hitting L2 (2 x 575 KB resident). 16 gathers + 3 plane loads + 3 stores.
// ---------------------------------------------------------------------------
struct F3 { float x, y, z; };

__device__ __forceinline__ F3 trilerp(const float4* __restrict__ L,
                                      float r, float g, float b)
{
    const float S = (float)(LUT_D - 1);
    const float xr = r * S, xg = g * S, xb = b * S;
    int ir = (int)floorf(xr); ir = ir < 0 ? 0 : (ir > LUT_D - 2 ? LUT_D - 2 : ir);
    int ig = (int)floorf(xg); ig = ig < 0 ? 0 : (ig > LUT_D - 2 ? LUT_D - 2 : ig);
    int ib = (int)floorf(xb); ib = ib < 0 ? 0 : (ib > LUT_D - 2 ? LUT_D - 2 : ib);
    const float fr = xr - (float)ir;
    const float fg = xg - (float)ig;
    const float fb = xb - (float)ib;

    const int base = (ib * LUT_D + ig) * LUT_D + ir;
    const float4 c000 = L[base];
    const float4 c001 = L[base + 1];
    const float4 c010 = L[base + LUT_D];
    const float4 c011 = L[base + LUT_D + 1];
    const float4 c100 = L[base + LUT_D * LUT_D];
    const float4 c101 = L[base + LUT_D * LUT_D + 1];
    const float4 c110 = L[base + LUT_D * LUT_D + LUT_D];
    const float4 c111 = L[base + LUT_D * LUT_D + LUT_D + 1];

    const float gr = 1.0f - fr, gg = 1.0f - fg, gb = 1.0f - fb;
    const float w000 = gb * gg * gr, w001 = gb * gg * fr;
    const float w010 = gb * fg * gr, w011 = gb * fg * fr;
    const float w100 = fb * gg * gr, w101 = fb * gg * fr;
    const float w110 = fb * fg * gr, w111 = fb * fg * fr;

    F3 o;
    o.x = w000 * c000.x + w001 * c001.x + w010 * c010.x + w011 * c011.x
        + w100 * c100.x + w101 * c101.x + w110 * c110.x + w111 * c111.x;
    o.y = w000 * c000.y + w001 * c001.y + w010 * c010.y + w011 * c011.y
        + w100 * c100.y + w101 * c101.y + w110 * c110.y + w111 * c111.y;
    o.z = w000 * c000.z + w001 * c001.z + w010 * c010.z + w011 * c011.z
        + w100 * c100.z + w101 * c101.z + w110 * c110.z + w111 * c111.z;
    return o;
}

__global__ __launch_bounds__(256) void lut_apply_fused(
    const float* __restrict__ gt,
    const float4* __restrict__ cl0,
    const float4* __restrict__ cl1,
    float* __restrict__ out, int hw)
{
    const int p = blockIdx.x * blockDim.x + threadIdx.x;
    if (p >= hw) return;
    const float r = gt[p];
    const float g = gt[hw + p];
    const float b = gt[2 * hw + p];
    const F3 s = trilerp(cl0, r, g, b);
    const F3 f = trilerp(cl1, s.x, s.y, s.z);
    out[p]          = f.x;
    out[hw + p]     = f.y;
    out[2 * hw + p] = f.z;
}

// ---------------------------------------------------------------------------
extern "C" void kernel_launch(void* const* d_in, const int* in_sizes, int n_in,
                              void* d_out, int out_size, void* d_ws, size_t ws_size,
                              hipStream_t stream)
{
    const float* gt  = (const float*)d_in[0];   // [3, H, W]
    const float* lut = (const float*)d_in[1];   // [2, 350, 3, 33, 33, 33]
    // d_in[2]/d_in[3] (L0/L1) are dead code in the reference.
    float* cl = (float*)d_ws;                   // [2][35937][4] f32 = 1.15 MB

    const int waves   = (TOTCOLS + 15) / 16;    // 13477 waves of 16 columns
    const int blocks1 = (waves + 7) / 8;        // 8 waves / 256-thread block
    lut_reduce_wmma<<<blocks1, 256, 0, stream>>>(lut, cl);

    const int hw      = in_sizes[0] / 3;        // 2160 * 3840
    const int blocks2 = (hw + 255) / 256;
    lut_apply_fused<<<blocks2, 256, 0, stream>>>(
        gt, (const float4*)cl, (const float4*)(cl + CL_STRIDE),
        (float*)d_out, hw);
}
